// LinearAttention_31473520345440
// MI455X (gfx1250) — matrix operand order repaired and checked by reading, not compiled, not run on hardware
//
#include <hip/hip_runtime.h>

typedef __bf16 bf16_t;
typedef __attribute__((ext_vector_type(16))) __bf16 v16bf;
typedef __attribute__((ext_vector_type(8)))  __bf16 v8bf;
typedef __attribute__((ext_vector_type(8)))  float  v8f;

#define B_ 4
#define S_ 4096
#define D_ 256
#define U_ 256
#define R_ 256
#define M_ (B_*S_)   // 16384 total rows

static __device__ __forceinline__ v8f wmma_bf16(v16bf a, v16bf b, v8f c) {
  // D = A(16x32 bf16) * B(32x16 bf16) + C(16x16 f32)
  return __builtin_amdgcn_wmma_f32_16x16x32_bf16(false, a, false, b, (short)0, c, false, false);
}

// Async DMA: 16 bytes per lane, global -> LDS, tracked by ASYNCcnt.
#define ASYNC_LD128(ldsoff, gptr, imm)                                        \
  asm volatile("global_load_async_to_lds_b128 %0, %1, off offset:" #imm      \
               :: "v"(ldsoff), "v"(gptr) : "memory")

// ---------------------------------------------------------------------------
// Pack fp32 weight matrices [256x256] into WMMA B-operand layout, bf16.
// Packed element ((rc*16+nt)*32 + lane)*16 + e = W[rc*32 + 16*(lane/16) + e, nt*16 + lane%16]
// ---------------------------------------------------------------------------
__global__ void __launch_bounds__(256)
pack_weights(const float* __restrict__ Wq, const float* __restrict__ Wk,
             const float* __restrict__ Wv, const float* __restrict__ Wr,
             bf16_t* __restrict__ pk) {
  int id   = blockIdx.x * 256 + threadIdx.x;     // 4 * 65536 total
  int e    = id & 15;
  int lane = (id >> 4) & 31;
  int nt   = (id >> 9) & 15;
  int rc   = (id >> 13) & 7;
  int m    = id >> 16;
  const float* W = (m == 0) ? Wq : (m == 1) ? Wk : (m == 2) ? Wv : Wr;
  int k = rc * 32 + ((lane >> 4) << 4) + e;
  int n = nt * 16 + (lane & 15);
  pk[id] = (bf16_t)W[k * U_ + n];
}

// ---------------------------------------------------------------------------
// Pass 1: Q = query*Wq+bq, K = value*Wk+bk, V = value*Wv+bv (V transposed).
// 3 waves per block, one projection each; 16 output rows per block.
// ---------------------------------------------------------------------------
__global__ void __launch_bounds__(96)
proj_kernel(const float* __restrict__ query, const float* __restrict__ value,
            const bf16_t* __restrict__ Wpk,
            const float* __restrict__ bq, const float* __restrict__ bk,
            const float* __restrict__ bv,
            bf16_t* __restrict__ Qbf, bf16_t* __restrict__ Kbf,
            bf16_t* __restrict__ Vt) {
  __shared__ __align__(32) bf16_t ldsT[16 * 16];   // only wave 2 uses it
  const int w    = threadIdx.x >> 5;
  const int lane = threadIdx.x & 31;
  const int ln   = lane & 15, lh = lane >> 4;
  const int m0   = blockIdx.x * 16;

  const float*  in   = (w == 0) ? query : value;
  const bf16_t* Wm   = Wpk + (size_t)w * 65536;
  const float*  bias = (w == 0) ? bq : (w == 1) ? bk : bv;

  v8f acc[16];
  #pragma unroll
  for (int ut = 0; ut < 16; ++ut) acc[ut] = (v8f){0,0,0,0,0,0,0,0};

  for (int rc = 0; rc < 8; ++rc) {
    const float* ap = in + (size_t)(m0 + ln) * D_ + rc * 32 + lh * 16;
    v16bf a;
    #pragma unroll
    for (int i = 0; i < 8; ++i) a[i]     = (bf16_t)ap[i];
    #pragma unroll
    for (int i = 0; i < 8; ++i) a[8 + i] = (bf16_t)ap[8 + i];
    #pragma unroll
    for (int ut = 0; ut < 16; ++ut) {
      v16bf bb = *(const v16bf*)(Wm + ((size_t)(rc * 16 + ut) * 32 + lane) * 16);
      acc[ut] = wmma_bf16(a, bb, acc[ut]);
    }
  }

  if (w < 2) {
    bf16_t* outp = (w == 0) ? Qbf : Kbf;
    #pragma unroll
    for (int ut = 0; ut < 16; ++ut) {
      float bb = bias[ut * 16 + ln];
      #pragma unroll
      for (int v = 0; v < 8; ++v) {
        int row = m0 + v + lh * 8;
        outp[(size_t)row * U_ + ut * 16 + ln] = (bf16_t)(acc[ut][v] + bb);
      }
    }
  } else {
    int b  = m0 >> 12;            // S_ == 4096
    int s0 = m0 & (S_ - 1);
    for (int ut = 0; ut < 16; ++ut) {
      float bb = bias[ut * 16 + ln];
      #pragma unroll
      for (int v = 0; v < 8; ++v)
        ldsT[(v + lh * 8) * 16 + ln] = (bf16_t)(acc[ut][v] + bb);
      asm volatile("s_wait_dscnt 0x0" ::: "memory");
      if (lane < 16) {
        v16bf col;
        #pragma unroll
        for (int s = 0; s < 16; ++s) col[s] = ldsT[s * 16 + lane];
        *(v16bf*)(Vt + ((size_t)b * U_ + ut * 16 + lane) * S_ + s0) = col;
      }
      asm volatile("s_wait_dscnt 0x0" ::: "memory");
    }
  }
}

// ---------------------------------------------------------------------------
// Pass 2: K_hat = cos(K * Wr + br), bf16 row-major output.
// ---------------------------------------------------------------------------
__global__ void __launch_bounds__(128)
khat_kernel(const bf16_t* __restrict__ Kbf, const bf16_t* __restrict__ Wrpk,
            const float* __restrict__ br, bf16_t* __restrict__ Khat) {
  const int w    = threadIdx.x >> 5;
  const int lane = threadIdx.x & 31;
  const int ln   = lane & 15, lh = lane >> 4;
  const int m0   = (blockIdx.x * 4 + w) * 16;

  v8f acc[16];
  #pragma unroll
  for (int ut = 0; ut < 16; ++ut) acc[ut] = (v8f){0,0,0,0,0,0,0,0};

  for (int rc = 0; rc < 8; ++rc) {
    const bf16_t* ap = Kbf + (size_t)(m0 + ln) * U_ + rc * 32 + lh * 16;
    union { v16bf v; v8bf h[2]; } a;
    a.h[0] = *(const v8bf*)(ap);
    a.h[1] = *(const v8bf*)(ap + 8);
    #pragma unroll
    for (int ut = 0; ut < 16; ++ut) {
      v16bf bb = *(const v16bf*)(Wrpk + ((size_t)(rc * 16 + ut) * 32 + lane) * 16);
      acc[ut] = wmma_bf16(a.v, bb, acc[ut]);
    }
  }
  #pragma unroll
  for (int ut = 0; ut < 16; ++ut) {
    float bb = br[ut * 16 + ln];
    #pragma unroll
    for (int v = 0; v < 8; ++v) {
      int row = m0 + v + lh * 8;
      Khat[(size_t)row * R_ + ut * 16 + ln] = (bf16_t)__cosf(acc[ut][v] + bb);
    }
  }
}

// ---------------------------------------------------------------------------
// Pass 3: flash attention. 4 waves/block, 16 query rows per wave.
// K_hat tiles: async-DMA'd into double-buffered LDS (ASYNCcnt).
// V B-operands: contiguous 32B/lane loads straight from L2-resident Vt.
// ---------------------------------------------------------------------------
__global__ void __launch_bounds__(128)
attn_kernel(const bf16_t* __restrict__ Qbf, const bf16_t* __restrict__ Khat,
            const bf16_t* __restrict__ Vt, float* __restrict__ out) {
  __shared__ __align__(32) bf16_t ldsK[2][32 * 256];   // 2 x 16 KB double buffer
  __shared__ __align__(32) bf16_t ldsP[4][16 * 32];    // per-wave P relayout
  const int tid  = threadIdx.x;
  const int w    = tid >> 5;
  const int lane = tid & 31;
  const int ln   = lane & 15, lh = lane >> 4;
  const int nq   = S_ / 64;
  const int b    = blockIdx.x / nq;
  const int q0   = (blockIdx.x % nq) * 64 + w * 16;

  // Q tile in registers, A-operand layout (64 VGPRs)
  v16bf qa[8];
  #pragma unroll
  for (int rc = 0; rc < 8; ++rc) {
    const bf16_t* qp = Qbf + ((size_t)b * S_ + q0 + ln) * U_ + rc * 32 + lh * 16;
    union { v16bf v; v8bf h[2]; } a;
    a.h[0] = *(const v8bf*)(qp);
    a.h[1] = *(const v8bf*)(qp + 8);
    qa[rc] = a.v;
  }

  v8f acc[16];
  #pragma unroll
  for (int ut = 0; ut < 16; ++ut) acc[ut] = (v8f){0,0,0,0,0,0,0,0};
  float mrow[8], lrow[8];
  #pragma unroll
  for (int v = 0; v < 8; ++v) { mrow[v] = -1e30f; lrow[v] = 0.f; }

  // staging addresses: 32 rows x 512B, 4 threads/row, 128B (8 x b128) per thread
  const int srow = tid >> 2, spart = tid & 3;
  const bf16_t* gK0 = Khat + ((size_t)b * S_ + srow) * R_ + spart * 64;

  // prologue: kick off DMA of tile 0 into buffer 0
  {
    const bf16_t* gp = gK0;
    unsigned lo = (unsigned)(uintptr_t)(&ldsK[0][srow * R_ + spart * 64]);
    ASYNC_LD128(lo, gp, 0);  ASYNC_LD128(lo, gp, 16);
    ASYNC_LD128(lo, gp, 32); ASYNC_LD128(lo, gp, 48);
    ASYNC_LD128(lo, gp, 64); ASYNC_LD128(lo, gp, 80);
    ASYNC_LD128(lo, gp, 96); ASYNC_LD128(lo, gp, 112);
  }

  for (int j = 0; j < S_ / 32; ++j) {
    const int key0 = j * 32;
    const bf16_t* kb = &ldsK[j & 1][0];

    asm volatile("s_wait_asynccnt 0x0" ::: "memory");
    __syncthreads();                       // tile j visible to all waves

    if (j + 1 < S_ / 32) {                 // DMA tile j+1 into the other buffer
      const bf16_t* gp = gK0 + (size_t)(key0 + 32) * R_;
      unsigned lo = (unsigned)(uintptr_t)(&ldsK[(j + 1) & 1][srow * R_ + spart * 64]);
      ASYNC_LD128(lo, gp, 0);  ASYNC_LD128(lo, gp, 16);
      ASYNC_LD128(lo, gp, 32); ASYNC_LD128(lo, gp, 48);
      ASYNC_LD128(lo, gp, 64); ASYNC_LD128(lo, gp, 80);
      ASYNC_LD128(lo, gp, 96); ASYNC_LD128(lo, gp, 112);
    }

    // scores: 16 q rows x 32 keys, f32 accum
    v8f s0 = (v8f){0,0,0,0,0,0,0,0}, s1 = (v8f){0,0,0,0,0,0,0,0};
    #pragma unroll
    for (int rc = 0; rc < 8; ++rc) {
      v16bf b0 = *(const v16bf*)(kb + ln * R_ + rc * 32 + lh * 16);
      v16bf b1 = *(const v16bf*)(kb + (16 + ln) * R_ + rc * 32 + lh * 16);
      s0 = wmma_bf16(qa[rc], b0, s0);
      s1 = wmma_bf16(qa[rc], b1, s1);
    }

    // online softmax (rows live per-(vgpr, lane-half); reduce over 16-lane groups)
    float corr[8];
    #pragma unroll
    for (int v = 0; v < 8; ++v) {
      float mv = fmaxf(s0[v], s1[v]);
      #pragma unroll
      for (int mk = 8; mk >= 1; mk >>= 1) mv = fmaxf(mv, __shfl_xor(mv, mk, 32));
      float mnew = fmaxf(mrow[v], mv);
      corr[v] = __expf(mrow[v] - mnew);
      mrow[v] = mnew;
      float p0 = __expf(s0[v] - mnew);
      float p1 = __expf(s1[v] - mnew);
      s0[v] = p0; s1[v] = p1;
      float rs = p0 + p1;
      #pragma unroll
      for (int mk = 8; mk >= 1; mk >>= 1) rs += __shfl_xor(rs, mk, 32);
      lrow[v] = lrow[v] * corr[v] + rs;
    }
    #pragma unroll
    for (int ut = 0; ut < 16; ++ut) {
      #pragma unroll
      for (int v = 0; v < 8; ++v) acc[ut][v] *= corr[v];
    }

    // P: C-layout -> A-layout via per-wave LDS (same-wave DS ops are in-order)
    bf16_t* pw = &ldsP[w][0];
    #pragma unroll
    for (int v = 0; v < 8; ++v) {
      int row = v + lh * 8;
      pw[row * 32 + ln]      = (bf16_t)s0[v];
      pw[row * 32 + 16 + ln] = (bf16_t)s1[v];
    }
    asm volatile("s_wait_dscnt 0x0" ::: "memory");
    union { v16bf v; v8bf h[2]; } pa;
    pa.h[0] = *(const v8bf*)(pw + ln * 32 + lh * 16);
    pa.h[1] = *(const v8bf*)(pw + ln * 32 + 8 + lh * 16);

    // acc += P * V ; V B-operand read directly from global (L2-resident),
    // contiguous 32B per lane thanks to the Vt[b][u][s] layout.
    const bf16_t* vb = Vt + ((size_t)b * U_ + ln) * S_ + key0 + lh * 16;
    #pragma unroll
    for (int ut = 0; ut < 16; ++ut) {
      v16bf bv = *(const v16bf*)(vb + (size_t)ut * 16 * S_);
      acc[ut] = wmma_bf16(pa.v, bv, acc[ut]);
    }
  }

  // epilogue: normalize and store fp32 output
  float inv[8];
  #pragma unroll
  for (int v = 0; v < 8; ++v) inv[v] = 1.0f / lrow[v];
  #pragma unroll
  for (int ut = 0; ut < 16; ++ut) {
    #pragma unroll
    for (int v = 0; v < 8; ++v) {
      int row = q0 + v + lh * 8;
      out[((size_t)b * S_ + row) * U_ + ut * 16 + ln] = acc[ut][v] * inv[v];
    }
  }
}

// ---------------------------------------------------------------------------
extern "C" void kernel_launch(void* const* d_in, const int* in_sizes, int n_in,
                              void* d_out, int out_size, void* d_ws, size_t ws_size,
                              hipStream_t stream) {
  (void)in_sizes; (void)n_in; (void)out_size; (void)ws_size;
  const float* query = (const float*)d_in[0];
  const float* value = (const float*)d_in[1];
  const float* Wq    = (const float*)d_in[2];
  const float* bq    = (const float*)d_in[3];
  const float* Wk    = (const float*)d_in[4];
  const float* bk    = (const float*)d_in[5];
  const float* Wv    = (const float*)d_in[6];
  const float* bv    = (const float*)d_in[7];
  const float* Wr    = (const float*)d_in[8];
  const float* br    = (const float*)d_in[9];
  float* out = (float*)d_out;

  char* ws = (char*)d_ws;
  bf16_t* Wpk  = (bf16_t*)(ws);                                  // 512 KB (4 packed weights)
  bf16_t* Qbf  = (bf16_t*)(ws + (size_t)(512 << 10));            // 8 MB
  bf16_t* Kbf  = (bf16_t*)(ws + (size_t)(512 << 10) + (size_t)(8  << 20));
  bf16_t* Khat = (bf16_t*)(ws + (size_t)(512 << 10) + (size_t)(16 << 20));
  bf16_t* Vt   = (bf16_t*)(ws + (size_t)(512 << 10) + (size_t)(24 << 20));

  pack_weights<<<1024, 256, 0, stream>>>(Wq, Wk, Wv, Wr, Wpk);
  proj_kernel<<<M_ / 16, 96, 0, stream>>>(query, value, Wpk, bq, bk, bv, Qbf, Kbf, Vt);
  khat_kernel<<<M_ / 64, 128, 0, stream>>>(Kbf, Wpk + 3 * 65536, br, Khat);
  attn_kernel<<<B_ * (S_ / 64), 128, 0, stream>>>(Qbf, Khat, Vt, out);
}
